// SelfAttention_89120571392174
// MI455X (gfx1250) — matrix operand order repaired and checked by reading, not compiled, not run on hardware
//
#include <hip/hip_runtime.h>
#include <hip/hip_bf16.h>

// ---------------------------------------------------------------------------
// Self-attention (N=4, L=2048, E=768, H=12, D=64), f32 in/out.
// bf16 WMMA (v_wmma_f32_16x16x32_bf16) for all 4 matmuls. Flash attention in
// the S^T orientation; K / V^T tiles staged into LDS with the CDNA5 Tensor
// Data Mover (TENSOR_LOAD_TO_LDS + s_wait_tensorcnt), double buffered, shared
// by all 8 waves of the workgroup.
// ---------------------------------------------------------------------------

#define EMBED 768
#define HEADS 12
#define HDIM  64
#define NBATCH 4
#define SEQ   2048
#define KTILE 32

typedef __bf16 bf16;
typedef __attribute__((ext_vector_type(16))) __bf16 v16bf;
typedef __attribute__((ext_vector_type(8)))  __bf16 v8bf;
typedef __attribute__((ext_vector_type(8)))  float  v8f;
typedef __attribute__((ext_vector_type(4)))  unsigned u32x4;
typedef __attribute__((ext_vector_type(8)))  int      i32x8;
typedef __attribute__((ext_vector_type(4)))  int      i32x4;

static __device__ __forceinline__ int lane_id() { return threadIdx.x & 31; }

static __device__ __forceinline__ v8f wmma_bf16(v16bf a, v16bf b, v8f c) {
  // (neg_a, A, neg_b, B, c_mod, C, reuse_a, reuse_b)
  return __builtin_amdgcn_wmma_f32_16x16x32_bf16(false, a, false, b, (short)0, c,
                                                 false, false);
}

// A fragment 16x32 (M x K) bf16 from f32 row-major memory.
// Lane l<16: row=l, K={0..15}; lane l>=16: row=l-16, K={16..31}.
static __device__ __forceinline__ v16bf load_a_f32(const float* __restrict__ base, int ld) {
  const int lane = lane_id();
  const float* p = base + (lane & 15) * ld + ((lane >> 4) << 4);
  v16bf a;
#pragma unroll
  for (int i = 0; i < 8; ++i) a[i] = (__bf16)p[i];
#pragma unroll
  for (int i = 0; i < 8; ++i) a[8 + i] = (__bf16)p[8 + i];
  return a;
}

// Same A fragment from bf16 row-major memory (global or LDS; two 16B loads).
static __device__ __forceinline__ v16bf load_a_bf16(const bf16* base, int ld) {
  const int lane = lane_id();
  const bf16* p = base + (lane & 15) * ld + ((lane >> 4) << 4);
  v8bf lo = *(const v8bf*)p;
  v8bf hi = *(const v8bf*)(p + 8);
  v16bf a;
#pragma unroll
  for (int i = 0; i < 8; ++i) { a[i] = lo[i]; a[8 + i] = hi[i]; }
  return a;
}

// B fragment 32x16 (K x N) bf16 from a K-contiguous (transposed) source:
// source row n holds the K-run of output column n. One 32B load per lane.
static __device__ __forceinline__ v16bf load_b_bf16(const bf16* __restrict__ base, int ld) {
  const int lane = lane_id();
  const bf16* p = base + (lane & 15) * ld + ((lane >> 4) << 4);
  return *(const v16bf*)p;
}

static __device__ __forceinline__ unsigned pack_bf16x2(float a, float b) {
  union { __bf16 h[2]; unsigned u; } u_;
  u_.h[0] = (__bf16)a; u_.h[1] = (__bf16)b;
  return u_.u;
}

// LDS byte offset of a __shared__ pointer: LDS aperture keeps the offset in
// the low 32 bits of the flat address.
static __device__ __forceinline__ unsigned lds_offset_of(const void* p) {
  return (unsigned)(unsigned long long)p;
}

// ---------------------------------------------------------------------------
// TDM: 2-D tile load Global -> LDS. D# packed per CDNA5 ISA 8.3/8.4:
//  g0: [1:0]=count=1, [63:32]=lds_addr, [120:64]=global_addr, [127:126]=type=2
//  g1: [17:16]=data_size(1 => 2B), [79:48]=tensor_dim0, [111:80]=tensor_dim1,
//      [127:112]=tile_dim0, [143:128]=tile_dim1, [159:144]=tile_dim2(0),
//      [207:160]=tensor_dim0_stride, [255:208]=tensor_dim1_stride(0)
// Groups 2/3 zero (dims 2..4 unused). Tracked with TENSORcnt.
// ---------------------------------------------------------------------------
static __device__ __forceinline__ void tdm_load_2d(unsigned lds_off, const void* gptr,
                                                   unsigned tile_d0, unsigned tile_d1,
                                                   unsigned tensor_d0, unsigned tensor_d1,
                                                   unsigned long long stride0) {
  const unsigned long long ga = (unsigned long long)gptr;
  u32x4 g0;
  g0[0] = 1u;                                              // count=1, user desc
  g0[1] = lds_off;                                         // lds_addr
  g0[2] = (unsigned)(ga & 0xFFFFFFFFu);                    // global_addr lo
  g0[3] = (unsigned)((ga >> 32) & 0x01FFFFFFu) | 0x80000000u; // addr hi | type=2
  i32x8 g1;
  g1[0] = (int)(1u << 16);                                 // data_size = 2 bytes
  g1[1] = (int)((tensor_d0 & 0xFFFFu) << 16);
  g1[2] = (int)((tensor_d0 >> 16) | ((tensor_d1 & 0xFFFFu) << 16));
  g1[3] = (int)((tensor_d1 >> 16) | (tile_d0 << 16));
  g1[4] = (int)tile_d1;                                    // tile_dim2 = 0
  g1[5] = (int)(stride0 & 0xFFFFFFFFull);
  g1[6] = (int)((stride0 >> 32) & 0xFFFFull);              // dim1_stride = 0
  g1[7] = 0;
  const i32x4 z4 = {0, 0, 0, 0};
#if defined(__clang_major__) && (__clang_major__ >= 23)
  const i32x8 z8 = {0, 0, 0, 0, 0, 0, 0, 0};
  __builtin_amdgcn_tensor_load_to_lds(g0, g1, z4, z4, z8, 0);
#else
  __builtin_amdgcn_tensor_load_to_lds(g0, g1, z4, z4, 0);
#endif
}

// ---------------------------------------------------------------------------
// Kernel 0a: Wq/Wk/Wv [64x64] f32 -> transposed bf16 (row e holds W[:,e]).
// ---------------------------------------------------------------------------
__global__ void prep_small(const float* __restrict__ Wq, const float* __restrict__ Wk,
                           const float* __restrict__ Wv,
                           bf16* __restrict__ WqT, bf16* __restrict__ WkT,
                           bf16* __restrict__ WvT) {
  for (int i = threadIdx.x; i < HDIM * HDIM; i += blockDim.x) {
    int d = i / HDIM, e = i % HDIM;
    WqT[e * HDIM + d] = (__bf16)Wq[i];
    WkT[e * HDIM + d] = (__bf16)Wk[i];
    WvT[e * HDIM + d] = (__bf16)Wv[i];
  }
}

// Kernel 0b: Wo [768x768] f32 -> transposed bf16.
__global__ void prep_wo(const float* __restrict__ Wo, bf16* __restrict__ WoT) {
  for (int i = blockIdx.x * blockDim.x + threadIdx.x; i < EMBED * EMBED;
       i += gridDim.x * blockDim.x) {
    int k = i / EMBED, n = i % EMBED;
    WoT[n * EMBED + k] = (__bf16)Wo[i];
  }
}

// ---------------------------------------------------------------------------
// Kernel 1: QKV projection. x viewed as [N*L*H, 64] row-major (d innermost).
// Q/K stored [n][h][l][d]; V stored transposed [n][h][d][l].
// ---------------------------------------------------------------------------
template <bool STORE_TRANSPOSED>
static __device__ __forceinline__ void proj_one(v16bf a0, v16bf a1,
                                                const bf16* __restrict__ WT,
                                                bf16* __restrict__ dst, int rbase) {
  const int lane = lane_id();
  const int rofs = (lane >> 4) << 3;
#pragma unroll
  for (int et = 0; et < 4; ++et) {
    v8f c = {};
    c = wmma_bf16(a0, load_b_bf16(WT + et * 16 * HDIM, HDIM), c);
    c = wmma_bf16(a1, load_b_bf16(WT + et * 16 * HDIM + 32, HDIM), c);
    const int e = et * 16 + (lane & 15);
#pragma unroll
    for (int j = 0; j < 8; ++j) {
      const int r = rbase + j + rofs;      // global (n,l,h) row
      const int h = r % HEADS;
      const int t = r / HEADS;
      const int l = t % SEQ;
      const int n = t / SEQ;
      if (STORE_TRANSPOSED)
        dst[((n * HEADS + h) * HDIM + e) * SEQ + l] = (__bf16)c[j];
      else
        dst[((n * HEADS + h) * SEQ + l) * HDIM + e] = (__bf16)c[j];
    }
  }
}

__global__ void __launch_bounds__(256) qkv_proj(const float* __restrict__ x,
                                                const bf16* __restrict__ WqT,
                                                const bf16* __restrict__ WkT,
                                                const bf16* __restrict__ WvT,
                                                bf16* __restrict__ Q,
                                                bf16* __restrict__ K,
                                                bf16* __restrict__ VT) {
  const int wave = (blockIdx.x * blockDim.x + threadIdx.x) >> 5;
  const int rbase = wave * 16;
  const float* xa = x + (long)rbase * HDIM;
  v16bf a0 = load_a_f32(xa, HDIM);        // d 0..31
  v16bf a1 = load_a_f32(xa + 32, HDIM);   // d 32..63
  proj_one<false>(a0, a1, WqT, Q, rbase);
  proj_one<false>(a0, a1, WkT, K, rbase);
  proj_one<true >(a0, a1, WvT, VT, rbase);
}

// ---------------------------------------------------------------------------
// Kernel 2: flash attention. Block = 8 waves sharing one (n,h); each wave owns
// a 16-query tile. Per 32-key step: TDM stages K tile [32x64] and V^T tile
// [64x32] into double-buffered LDS (wave 0 issues, s_wait_tensorcnt + barrier
// publishes); every wave then runs 8 WMMA on ds_load fragments.
// ---------------------------------------------------------------------------
__global__ void __launch_bounds__(256) attn(const bf16* __restrict__ Q,
                                            const bf16* __restrict__ K,
                                            const bf16* __restrict__ VT,
                                            bf16* __restrict__ AO) {
  __shared__ bf16 Kbuf[2][KTILE * HDIM];   // [key][d]   4KB x2
  __shared__ bf16 Vbuf[2][HDIM * KTILE];   // [d][key]   4KB x2

  const int tid = threadIdx.x;
  const int waveid = tid >> 5;
  const int lane = tid & 31;
  const int nh = blockIdx.x >> 4;          // 0..47 : (n*HEADS + h)
  const int qt = (blockIdx.x & 15) * 8 + waveid;  // 0..127 query tile
  const int h = nh % HEADS, n = nh / HEADS;

  const bf16* Qb = Q + ((long)nh * SEQ + qt * 16) * HDIM;
  const bf16* Kb = K + (long)nh * SEQ * HDIM;
  const bf16* Vb = VT + (long)nh * HDIM * SEQ;

  const unsigned kl[2] = {lds_offset_of(Kbuf[0]), lds_offset_of(Kbuf[1])};
  const unsigned vl[2] = {lds_offset_of(Vbuf[0]), lds_offset_of(Vbuf[1])};

  // Resident Q^T B-fragments (col = query, K-dim = d).
  const v16bf bq0 = load_b_bf16(Qb, HDIM);       // d 0..31
  const v16bf bq1 = load_b_bf16(Qb + 32, HDIM);  // d 32..63

  // Prologue: stage the first K / V^T tiles into buffer 0.
  if (waveid == 0) {
    tdm_load_2d(kl[0], Kb, /*tile*/ HDIM, KTILE, /*tensor*/ HDIM, SEQ, HDIM);
    tdm_load_2d(vl[0], Vb, /*tile*/ KTILE, HDIM, /*tensor*/ SEQ, HDIM, SEQ);
  }

  v8f o[4] = {v8f{}, v8f{}, v8f{}, v8f{}};       // O^T accum, d-tiles 0..3
  float mrun = -1e30f, lsum = 0.0f;
  const float sc = 0.03608439182435161f;         // 1/sqrt(EMBED)
  const bool lohalf = (lane < 16);

  int buf = 0;
  for (int kb = 0; kb < SEQ; kb += KTILE, buf ^= 1) {
    __syncthreads();                 // back buffer free (prev compute done)
    if (waveid == 0) {
      if (kb + KTILE < SEQ) {        // prefetch next pair into back buffer
        tdm_load_2d(kl[buf ^ 1], Kb + (long)(kb + KTILE) * HDIM,
                    HDIM, KTILE, HDIM, SEQ, HDIM);
        tdm_load_2d(vl[buf ^ 1], Vb + (kb + KTILE),
                    KTILE, HDIM, SEQ, HDIM, SEQ);
        __builtin_amdgcn_s_wait_tensorcnt(2);   // front pair complete
      } else {
        __builtin_amdgcn_s_wait_tensorcnt(0);
      }
    }
    __syncthreads();                 // front buffer published to all waves
    const bf16* Kt = Kbuf[buf];
    const bf16* Vt = Vbuf[buf];

    // ---- S^T for 32 keys: two 16x16 C fragments (from LDS) --------------
    v8f s0 = {}, s1 = {};
    s0 = wmma_bf16(load_a_bf16(Kt, HDIM), bq0, s0);
    s0 = wmma_bf16(load_a_bf16(Kt + 32, HDIM), bq1, s0);
    s1 = wmma_bf16(load_a_bf16(Kt + 16 * HDIM, HDIM), bq0, s1);
    s1 = wmma_bf16(load_a_bf16(Kt + 16 * HDIM + 32, HDIM), bq1, s1);

    // ---- online softmax (per-lane = per-query stats) --------------------
    float t0[8], t1[8];
    float tmax = -1e30f;
#pragma unroll
    for (int j = 0; j < 8; ++j) {
      t0[j] = s0[j] * sc;
      t1[j] = s1[j] * sc;
      tmax = fmaxf(tmax, fmaxf(t0[j], t1[j]));
    }
    tmax = fmaxf(tmax, __shfl_xor(tmax, 16, 32));
    const float mnew = fmaxf(mrun, tmax);
    const float corr = __expf(mrun - mnew);
    mrun = mnew;
    lsum *= corr;
#pragma unroll
    for (int j = 0; j < 8; ++j) {
      t0[j] = __expf(t0[j] - mnew);
      t1[j] = __expf(t1[j] - mnew);
      lsum += t0[j] + t1[j];
    }
#pragma unroll
    for (int tt = 0; tt < 4; ++tt)
#pragma unroll
      for (int j = 0; j < 8; ++j) o[tt][j] *= corr;

    // ---- build P^T B-fragment (32 keys x 16 queries) --------------------
    union { unsigned u[8]; v16bf v; } bp;
#pragma unroll
    for (int t = 0; t < 4; ++t) {
      const unsigned pk0 = pack_bf16x2(t0[2 * t], t0[2 * t + 1]); // keys {0..7}/{8..15}
      const unsigned pk1 = pack_bf16x2(t1[2 * t], t1[2 * t + 1]); // keys {16..23}/{24..31}
      const unsigned x0 = __shfl_xor(pk0, 16, 32);
      const unsigned x1 = __shfl_xor(pk1, 16, 32);
      bp.u[t]     = lohalf ? pk0 : x1;
      bp.u[4 + t] = lohalf ? x0 : pk1;
    }

    // ---- O^T accumulation: A = V^T rows from LDS ------------------------
#pragma unroll
    for (int tt = 0; tt < 4; ++tt)
      o[tt] = wmma_bf16(load_a_bf16(Vt + tt * 16 * KTILE, KTILE), bp.v, o[tt]);
  }

  lsum += __shfl_xor(lsum, 16, 32);
  const float inv = 1.0f / lsum;

  // ---- store O^T -> AO[n][l][h*64+d] (bf16) -----------------------------
  const int q = qt * 16 + (lane & 15);
  const int dofs = (lane >> 4) << 3;
#pragma unroll
  for (int tt = 0; tt < 4; ++tt)
#pragma unroll
    for (int j = 0; j < 8; ++j) {
      const int d = tt * 16 + dofs + j;
      AO[((long)(n * SEQ + q)) * EMBED + h * HDIM + d] = (__bf16)(o[tt][j] * inv);
    }
}

// ---------------------------------------------------------------------------
// Kernel 3: out = AO[8192x768] @ Wo + bo, f32 output. Wave: 16x64 tile.
// ---------------------------------------------------------------------------
__global__ void __launch_bounds__(256) out_proj(const bf16* __restrict__ AO,
                                                const bf16* __restrict__ WoT,
                                                const float* __restrict__ bo,
                                                float* __restrict__ out) {
  const int wave = (blockIdx.x * blockDim.x + threadIdx.x) >> 5;
  const int lane = lane_id();
  const int ntile = wave % (EMBED / 64);   // 12 column groups of 64
  const int mtile = wave / (EMBED / 64);   // 512 row tiles of 16
  const bf16* Ab = AO + (long)mtile * 16 * EMBED;

  v8f c[4] = {v8f{}, v8f{}, v8f{}, v8f{}};
  for (int kb = 0; kb < EMBED; kb += 32) {
    const v16bf a = load_a_bf16(Ab + kb, EMBED);
#pragma unroll
    for (int et = 0; et < 4; ++et) {
      const v16bf b = load_b_bf16(WoT + (long)(ntile * 64 + et * 16) * EMBED + kb, EMBED);
      c[et] = wmma_bf16(a, b, c[et]);
    }
  }

  const int rofs = (lane >> 4) << 3;
#pragma unroll
  for (int et = 0; et < 4; ++et) {
    const int col = ntile * 64 + et * 16 + (lane & 15);
    const float bias = bo[col];
#pragma unroll
    for (int j = 0; j < 8; ++j) {
      const int row = mtile * 16 + j + rofs;
      out[(long)row * EMBED + col] = c[et][j] + bias;
    }
  }
}

// ---------------------------------------------------------------------------
// Workspace layout (bytes). All regions fully written before read; ~51.5 MB.
// ---------------------------------------------------------------------------
#define WS_WQT 0u
#define WS_WKT 8192u
#define WS_WVT 16384u
#define WS_WOT 24576u                       // 768*768*2 = 1179648
#define WS_Q   1204224u                     // 4*12*2048*64*2 = 12582912
#define WS_K   13787136u
#define WS_VT  26370048u
#define WS_AO  38952960u                    // end 51535872

extern "C" void kernel_launch(void* const* d_in, const int* in_sizes, int n_in,
                              void* d_out, int out_size, void* d_ws, size_t ws_size,
                              hipStream_t stream) {
  const float* x  = (const float*)d_in[0];
  const float* Wq = (const float*)d_in[1];
  const float* Wk = (const float*)d_in[2];
  const float* Wv = (const float*)d_in[3];
  const float* Wo = (const float*)d_in[4];
  const float* bo = (const float*)d_in[5];
  float* out = (float*)d_out;
  char* ws = (char*)d_ws;

  bf16* WqT = (bf16*)(ws + WS_WQT);
  bf16* WkT = (bf16*)(ws + WS_WKT);
  bf16* WvT = (bf16*)(ws + WS_WVT);
  bf16* WoT = (bf16*)(ws + WS_WOT);
  bf16* Qd  = (bf16*)(ws + WS_Q);
  bf16* Kd  = (bf16*)(ws + WS_K);
  bf16* VTd = (bf16*)(ws + WS_VT);
  bf16* AOd = (bf16*)(ws + WS_AO);

  prep_small<<<1, 256, 0, stream>>>(Wq, Wk, Wv, WqT, WkT, WvT);
  prep_wo<<<576, 256, 0, stream>>>(Wo, WoT);
  // N*L*H rows / 16 per wave / 8 waves per block = 768 blocks
  qkv_proj<<<768, 256, 0, stream>>>(x, WqT, WkT, WvT, Qd, Kd, VTd);
  // 48 (n,h) x 16 query groups = 768 blocks of 8 waves
  attn<<<768, 256, 0, stream>>>(Qd, Kd, VTd, AOd);
  // (N*L/16)*(E/64) = 6144 waves / 8 = 768 blocks
  out_proj<<<768, 256, 0, stream>>>(AOd, WoT, bo, out);
}
